// embedding_fusion_68195490726508
// MI455X (gfx1250) — compile-verified, matrix operand order checked by reading, not run on hardware
//
#include <hip/hip_runtime.h>

// ---------------------------------------------------------------------------
// AAM-softmax loss, fused. MI455X (gfx1250), wave32, bf16 WMMA w/ f32 accum.
// ---------------------------------------------------------------------------

#define B_DIM   4096
#define IN_DIM  1024
#define P_DIM   256
#define C_DIM   5994
#define C_PAD   6000      // padded to a multiple of 16 for WMMA tiling

#define COS_M_C  0.98006657784124163f     // cos(0.2)
#define SIN_M_C  0.19866933079506122f     // sin(0.2)
#define TH_C    (-0.98006657784124163f)   // cos(pi - 0.2)
#define MM_C     0.03973386631901224f     // sin(pi - 0.2) * 0.2
#define S_SCALE  30.0f
#define BIGNEG  (-1.0e30f)

typedef unsigned short u16;
typedef __attribute__((ext_vector_type(16))) __bf16 v16bf;
typedef __attribute__((ext_vector_type(8)))  float  v8f;

__device__ __forceinline__ u16 f32_to_bf16(float f) {
  union { float f; unsigned int u; } c; c.f = f;
  unsigned int x = c.u;
  x += 0x7fffu + ((x >> 16) & 1u);   // round-to-nearest-even
  return (u16)(x >> 16);
}

// Load one 16x32 bf16 fragment (K-contiguous memory, leading dim `ld`).
// Per-lane layout per CDNA5 ISA: row = lane&15, K-half = (lane>>4)*8,
// elements 0..7 -> K = kb..kb+7, elements 8..15 -> K = kb+16..kb+23.
__device__ __forceinline__ v16bf load_frag_bf16(const u16* __restrict__ base,
                                                int row0, int ld, int kk, int lane) {
  const int r  = lane & 15;
  const int kb = (lane >> 4) << 3;
  const u16* p = base + (size_t)(row0 + r) * ld + kk + kb;
  union { v16bf v; uint4 u[2]; } f;
  f.u[0] = *reinterpret_cast<const uint4*>(p);
  f.u[1] = *reinterpret_cast<const uint4*>(p + 16);
  return f.v;
}

// ---------------------------------------------------------------------------
// 1) f32 -> bf16 conversion of x and W_fc
// ---------------------------------------------------------------------------
__global__ __launch_bounds__(256)
void convert_kernel(const float* __restrict__ x, const float* __restrict__ wfc,
                    u16* __restrict__ xb, u16* __restrict__ wfcb) {
  const int NX = B_DIM * IN_DIM;
  const int NW = P_DIM * IN_DIM;
  for (int i = blockIdx.x * 256 + threadIdx.x; i < NX + NW; i += gridDim.x * 256) {
    if (i < NX) xb[i] = f32_to_bf16(x[i]);
    else        wfcb[i - NX] = f32_to_bf16(wfc[i - NX]);
  }
}

// ---------------------------------------------------------------------------
// 2) e = x @ W_fc^T + b ; row-L2-normalize ; write bf16.
//    One block = 16 batch rows x all 256 cols. 8 waves x two 16x16 n-tiles.
// ---------------------------------------------------------------------------
__global__ __launch_bounds__(256)
void fc_norm_kernel(const u16* __restrict__ xb, const u16* __restrict__ wfcb,
                    const float* __restrict__ bfc, u16* __restrict__ enb) {
  __shared__ float etile[16][264];   // +8 pad vs bank conflicts
  const int m0   = blockIdx.x * 16;
  const int wave = threadIdx.x >> 5;
  const int lane = threadIdx.x & 31;
  const int n0   = wave * 32;

  v8f acc0 = {};
  v8f acc1 = {};
  for (int kk = 0; kk < IN_DIM; kk += 32) {
    v16bf a  = load_frag_bf16(xb,   m0,      IN_DIM, kk, lane);
    v16bf b0 = load_frag_bf16(wfcb, n0,      IN_DIM, kk, lane);
    v16bf b1 = load_frag_bf16(wfcb, n0 + 16, IN_DIM, kk, lane);
    acc0 = __builtin_amdgcn_wmma_f32_16x16x32_bf16(false, a, false, b0, (short)0, acc0, false, false);
    acc1 = __builtin_amdgcn_wmma_f32_16x16x32_bf16(false, a, false, b1, (short)0, acc1, false, false);
  }

  // C/D layout: M = r + 8*(lane>>4), N = lane&15 within the 16x16 tile.
  const int cl = lane & 15;
  const int mh = (lane >> 4) << 3;
  const float bias0 = bfc[n0 + cl];
  const float bias1 = bfc[n0 + 16 + cl];
#pragma unroll
  for (int r = 0; r < 8; ++r) {
    etile[mh + r][n0 + cl]      = acc0[r] + bias0;
    etile[mh + r][n0 + 16 + cl] = acc1[r] + bias1;
  }
  __syncthreads();

  // L2 normalize: each wave owns 2 rows; 8 cols per lane.
#pragma unroll
  for (int rr = 0; rr < 2; ++rr) {
    const int m = wave * 2 + rr;
    float v[8];
    float ss = 0.0f;
#pragma unroll
    for (int j = 0; j < 8; ++j) {
      v[j] = etile[m][lane + j * 32];
      ss += v[j] * v[j];
    }
#pragma unroll
    for (int off = 16; off; off >>= 1) ss += __shfl_xor(ss, off, 32);
    const float scale = 1.0f / fmaxf(sqrtf(ss), 1e-12f);
    u16* dst = enb + (size_t)(m0 + m) * P_DIM;
#pragma unroll
    for (int j = 0; j < 8; ++j) dst[lane + j * 32] = f32_to_bf16(v[j] * scale);
  }
}

// ---------------------------------------------------------------------------
// 3) Row-L2-normalize W_aam -> bf16, zero-pad rows [5994, 6000).
//    One wave per row.
// ---------------------------------------------------------------------------
__global__ __launch_bounds__(256)
void waam_norm_kernel(const float* __restrict__ W, u16* __restrict__ wab) {
  const int wave = threadIdx.x >> 5;
  const int lane = threadIdx.x & 31;
  const int row  = blockIdx.x * 8 + wave;
  if (row >= C_PAD) return;
  u16* dst = wab + (size_t)row * P_DIM;
  if (row >= C_DIM) {
#pragma unroll
    for (int j = 0; j < 8; ++j) dst[lane + j * 32] = 0;
    return;
  }
  const float* src = W + (size_t)row * P_DIM;
  float v[8];
  float ss = 0.0f;
#pragma unroll
  for (int j = 0; j < 8; ++j) {
    v[j] = src[lane + j * 32];
    ss += v[j] * v[j];
  }
#pragma unroll
  for (int off = 16; off; off >>= 1) ss += __shfl_xor(ss, off, 32);
  const float scale = 1.0f / fmaxf(sqrtf(ss), 1e-12f);
#pragma unroll
  for (int j = 0; j < 8; ++j) dst[lane + j * 32] = f32_to_bf16(v[j] * scale);
}

// ---------------------------------------------------------------------------
// 4) Fused cosine GEMM + angular margin + online log-softmax -> per-row NLL.
//    One block = 16 batch rows vs all C classes. A fragments live in VGPRs.
// ---------------------------------------------------------------------------
__global__ __launch_bounds__(256)
void aam_softmax_kernel(const u16* __restrict__ enb, const u16* __restrict__ wab,
                        const int* __restrict__ label, float* __restrict__ nll) {
  __shared__ int   sl[16];
  __shared__ float smx[8][16];
  __shared__ float ssm[8][16];
  __shared__ float stg[8][16];

  const int m0   = blockIdx.x * 16;
  const int wave = threadIdx.x >> 5;
  const int lane = threadIdx.x & 31;
  if (threadIdx.x < 16) sl[threadIdx.x] = label[m0 + threadIdx.x];
  __syncthreads();

  const int cl = lane & 15;
  const int mh = (lane >> 4) << 3;
  int lab[8];
#pragma unroll
  for (int r = 0; r < 8; ++r) lab[r] = sl[mh + r];

  // All 8 A fragments (16 rows x K=256) pinned in registers for the block.
  v16bf afr[8];
#pragma unroll
  for (int k = 0; k < 8; ++k) afr[k] = load_frag_bf16(enb, m0, P_DIM, k * 32, lane);

  float mx[8], sm[8], tg[8];
#pragma unroll
  for (int r = 0; r < 8; ++r) { mx[r] = BIGNEG; sm[r] = 0.0f; tg[r] = BIGNEG; }

  const int NTILES = C_PAD / 16;   // 375
  for (int t = wave; t < NTILES; t += 8) {
    const int n0 = t * 16;
    v8f acc = {};
#pragma unroll
    for (int k = 0; k < 8; ++k) {
      v16bf b = load_frag_bf16(wab, n0, P_DIM, k * 32, lane);
      acc = __builtin_amdgcn_wmma_f32_16x16x32_bf16(false, afr[k], false, b, (short)0, acc, false, false);
    }
    const int n = n0 + cl;
    if (n < C_DIM) {
#pragma unroll
      for (int r = 0; r < 8; ++r) {
        const float c = acc[r];
        float logit;
        if (n == lab[r]) {
          float s2 = 1.0f - c * c;
          s2 = fminf(fmaxf(s2, 0.0f), 1.0f);
          const float sine = sqrtf(s2);
          float phi = c * COS_M_C - sine * SIN_M_C;
          phi = ((c - TH_C) > 0.0f) ? phi : (c - MM_C);
          logit = S_SCALE * phi;
          tg[r] = logit;
        } else {
          logit = S_SCALE * c;
        }
        if (logit > mx[r]) {
          sm[r] = sm[r] * __expf(mx[r] - logit) + 1.0f;
          mx[r] = logit;
        } else {
          sm[r] += __expf(logit - mx[r]);
        }
      }
    }
  }

  // Reduce (max, sumexp, target) across the 16 lanes of each half-wave.
#pragma unroll
  for (int r = 0; r < 8; ++r) {
#pragma unroll
    for (int off = 1; off < 16; off <<= 1) {
      const float om = __shfl_xor(mx[r], off, 32);
      const float os = __shfl_xor(sm[r], off, 32);
      const float nm = fmaxf(mx[r], om);
      sm[r] = sm[r] * __expf(mx[r] - nm) + os * __expf(om - nm);
      mx[r] = nm;
      tg[r] = fmaxf(tg[r], __shfl_xor(tg[r], off, 32));
    }
  }
  if (cl == 0) {
#pragma unroll
    for (int r = 0; r < 8; ++r) {
      smx[wave][mh + r] = mx[r];
      ssm[wave][mh + r] = sm[r];
      stg[wave][mh + r] = tg[r];
    }
  }
  __syncthreads();

  // Merge 8 wave-partials per row; emit NLL.
  if (threadIdx.x < 16) {
    float m = BIGNEG, s = 0.0f, tt = BIGNEG;
    for (int w = 0; w < 8; ++w) {
      const float m2 = smx[w][threadIdx.x];
      const float s2 = ssm[w][threadIdx.x];
      const float nm = fmaxf(m, m2);
      s = s * __expf(m - nm) + s2 * __expf(m2 - nm);
      m = nm;
      tt = fmaxf(tt, stg[w][threadIdx.x]);
    }
    nll[m0 + threadIdx.x] = m + logf(s) - tt;
  }
}

// ---------------------------------------------------------------------------
// 5) Deterministic mean over 4096 NLL values -> scalar loss.
// ---------------------------------------------------------------------------
__global__ __launch_bounds__(256)
void mean_kernel(const float* __restrict__ nll, float* __restrict__ out) {
  __shared__ float red[256];
  float s = 0.0f;
  for (int i = threadIdx.x; i < B_DIM; i += 256) s += nll[i];
  red[threadIdx.x] = s;
  __syncthreads();
  for (int st = 128; st; st >>= 1) {
    if (threadIdx.x < st) red[threadIdx.x] += red[threadIdx.x + st];
    __syncthreads();
  }
  if (threadIdx.x == 0) out[0] = red[0] * (1.0f / (float)B_DIM);
}

// ---------------------------------------------------------------------------
extern "C" void kernel_launch(void* const* d_in, const int* in_sizes, int n_in,
                              void* d_out, int out_size, void* d_ws, size_t ws_size,
                              hipStream_t stream) {
  const float* x     = (const float*)d_in[0];
  const int*   label = (const int*)d_in[1];
  const float* Wfc   = (const float*)d_in[2];
  const float* bfc   = (const float*)d_in[3];
  const float* Waam  = (const float*)d_in[4];
  float* out = (float*)d_out;

  char* ws = (char*)d_ws;
  size_t off = 0;
  u16* xb   = (u16*)(ws + off); off += (size_t)B_DIM * IN_DIM * 2;   // 8 MiB
  u16* wfcb = (u16*)(ws + off); off += (size_t)P_DIM * IN_DIM * 2;   // 0.5 MiB
  u16* enb  = (u16*)(ws + off); off += (size_t)B_DIM * P_DIM * 2;    // 2 MiB
  u16* wab  = (u16*)(ws + off); off += (size_t)C_PAD * P_DIM * 2;    // ~3 MiB
  float* nll = (float*)(ws + off);                                    // 16 KiB

  convert_kernel<<<4096, 256, 0, stream>>>(x, Wfc, xb, wfcb);
  waam_norm_kernel<<<C_PAD / 8, 256, 0, stream>>>(Waam, wab);
  fc_norm_kernel<<<B_DIM / 16, 256, 0, stream>>>(xb, wfcb, bfc, enb);
  aam_softmax_kernel<<<B_DIM / 16, 256, 0, stream>>>(enb, wab, label, nll);
  mean_kernel<<<1, 256, 0, stream>>>(nll, out);
}
